// CliffordAdaptiveGraphAttention_82592221102877
// MI455X (gfx1250) — compile-verified, matrix operand order checked by reading, not run on hardware
//
#include <hip/hip_runtime.h>
#include <hip/hip_bf16.h>
#include <math.h>

typedef __attribute__((ext_vector_type(16))) _Float16 v16h;
typedef __attribute__((ext_vector_type(8)))  float    v8f;

#define ROWS 64
#define TPB  256

// f16-half offsets of the pre-transposed weights inside d_ws
#define WS_NP   0
#define WS_EP   16384
#define WS_IM1  32768     /* [128][256] */
#define WS_IM2  65536
#define WS_MP1  81920
#define WS_MP2  98304     /* [16][128]  */
#define WS_NTC  100352    /* [16][128]  */
#define WS_Q    102400
#define WS_K    118784
#define WS_V    135168
#define WS_OUT  151552
#define WS_TOTAL 167936   /* halfs = 335872 bytes */

// ---------------------------------------------------------------------------
// Async weight staging: GLOBAL_LOAD_ASYNC_TO_LDS_B128 (ASYNCcnt-tracked DMA,
// no VGPR round trip).  DS address = low 32 bits of the generic LDS pointer
// (ISA 10.2: LDS aperture uses addr[31:0]).
// ---------------------------------------------------------------------------
__device__ __forceinline__ void issue_w(const _Float16* ws_src, _Float16* lds_dst, int halfs) {
    uint32_t base = (uint32_t)(uintptr_t)lds_dst;
    for (int off = threadIdx.x * 8; off < halfs; off += TPB * 8) {
        asm volatile("global_load_async_to_lds_b128 %0, %1, off"
                     :: "v"(base + (uint32_t)(off * 2)),
                        "v"((const void*)(ws_src + off))
                     : "memory");
    }
}
__device__ __forceinline__ void wait_w() {
    asm volatile("s_wait_asynccnt 0x0" ::: "memory");
    __syncthreads();
}

// Prefetch a global region into cache (global_prefetch_b8 path)
__device__ __forceinline__ void prefetch_g(const _Float16* p, int halfs) {
    for (int i = threadIdx.x * 64; i < halfs; i += TPB * 64)
        __builtin_prefetch(p + i, 0, 1);
}

// ---------------------------------------------------------------------------
// Fragment loader per CDNA5 ISA 7.12.2 (wave32):
//   16-bit A 16x32: lanes 0-15 -> row M=l, K {0..7,16..23}; lanes 16-31 -> K {8..15,24..31}
// ---------------------------------------------------------------------------
__device__ __forceinline__ v16h load_frag(const _Float16* base, int stride, int k0, int lane) {
    int l  = lane & 15;
    int hi = (lane >> 4) & 1;
    const _Float16* p = base + l * stride + k0 + hi * 8;
    v16h r;
#pragma unroll
    for (int i = 0; i < 8; ++i) { r[i] = p[i]; r[i + 8] = p[i + 16]; }
    return r;
}

// Convert a 64x128 f32 tile from global into f16 LDS (zero-pad OOB rows)
__device__ __forceinline__ void cvt_tile(const float* g, _Float16* dst, int base, int B) {
    for (int i = threadIdx.x; i < ROWS * 128; i += TPB) {
        int r = i >> 7;
        float v = (base + r < B) ? g[(size_t)(base + r) * 128 + (i & 127)] : 0.f;
        dst[i] = (_Float16)v;
    }
}

// 64 x N x K GEMM via v_wmma_f32_16x16x32_f16.
// ACT: 0=none 1=relu 2=tanh.  F16OUT: true -> f16 LDS (stride so), false -> f32 LDS (stride N)
template <int ACT, bool F16OUT>
__device__ void gemm_ws(const _Float16* X, int sx, const _Float16* Wt, int K, int N,
                        const float* bias, _Float16* outh, int so, float* outf) {
    int lane = threadIdx.x & 31;
    int wave = threadIdx.x >> 5;
    int tiles = (ROWS / 16) * (N / 16);
    for (int t = wave; t < tiles; t += TPB / 32) {
        int tm = t & 3;
        int tn = t >> 2;
        v8f acc = {};
        for (int k0 = 0; k0 < K; k0 += 32) {
            v16h a = load_frag(X + tm * 16 * sx, sx, k0, lane);
            v16h b = load_frag(Wt + tn * 16 * K, K, k0, lane);
            acc = __builtin_amdgcn_wmma_f32_16x16x32_f16(false, a, false, b,
                                                         (short)0, acc, false, false);
        }
        int l = lane & 15, hi = lane >> 4;
        int col = tn * 16 + l;
        float bv = bias[col];
#pragma unroll
        for (int g = 0; g < 8; ++g) {
            int row = tm * 16 + g + hi * 8;
            float v = acc[g] + bv;
            if (ACT == 1) v = fmaxf(v, 0.f);
            if (ACT == 2) v = tanhf(v);
            if (F16OUT) outh[row * so + col] = (_Float16)v;
            else        outf[row * N + col] = v;
        }
    }
}

// Final GEMM with fused  + out_b + cto_b + ip@cto_W(LDS)  epilogue, f32 -> HBM
__device__ void gemm_final(const _Float16* X, const _Float16* Wt,
                           const float* out_b, const float* f_cto, const float* cto_b,
                           const float* f_ip, float* out_g, int base, bool full, int B) {
    int lane = threadIdx.x & 31;
    int wave = threadIdx.x >> 5;
    for (int t = wave; t < 32; t += 8) {
        int tm = t & 3, tn = t >> 2;
        v8f acc = {};
        for (int k0 = 0; k0 < 128; k0 += 32) {
            v16h a = load_frag(X + tm * 16 * 128, 128, k0, lane);
            v16h b = load_frag(Wt + tn * 16 * 128, 128, k0, lane);
            acc = __builtin_amdgcn_wmma_f32_16x16x32_f16(false, a, false, b,
                                                         (short)0, acc, false, false);
        }
        int l = lane & 15, hi = lane >> 4;
        int col = tn * 16 + l;
        float bsum = out_b[col] + cto_b[col];
        float res[8];
#pragma unroll
        for (int g = 0; g < 8; ++g) {
            int row = tm * 16 + g + hi * 8;
            float v = acc[g] + bsum;
            const float* ipr = f_ip + row * 16;
#pragma unroll
            for (int k = 0; k < 16; ++k) v += ipr[k] * f_cto[k * 128 + col];
            res[g] = v;
        }
        if (full) {
#pragma unroll
            for (int g = 0; g < 8; ++g) {
                int row = tm * 16 + g + hi * 8;
                out_g[(size_t)(base + row) * 128 + col] = res[g];
            }
        } else {
#pragma unroll
            for (int g = 0; g < 8; ++g) {
                int row = tm * 16 + g + hi * 8;
                if (base + row < B) out_g[(size_t)(base + row) * 128 + col] = res[g];
            }
        }
    }
}

__device__ __forceinline__ float softplus_f(float x) {
    return (x > 20.f) ? x : log1pf(expf(x));
}

// ---------------------------------------------------------------------------
// Prep pass: W[k*N+n] (f32) -> dst[n*K+k] (f16) so main kernel async-copies
// ready-to-use transposed f16 weights.
// ---------------------------------------------------------------------------
__global__ void wconv_kernel(const float* __restrict__ W, _Float16* __restrict__ dst,
                             int K, int N) {
    int idx = blockIdx.x * blockDim.x + threadIdx.x;
    if (idx >= K * N) return;
    int k = idx / N, n = idx - k * N;
    dst[n * K + k] = (_Float16)W[idx];
}

// ---------------------------------------------------------------------------
__global__ void __launch_bounds__(TPB)
caga_fused_kernel(const float* src, const float* dst, const float* edge,
                  const _Float16* ws,
                  const float* np_b, const float* ep_b, const float* im1_b,
                  const float* im2_b, const float* mp1_b, const float* mp2_b,
                  const float* q_b, const float* k_b, const float* v_b,
                  const float* ntc_b, const float* cto_W, const float* cto_b,
                  const float* out_b, float* out, int B) {
    extern __shared__ char smem[];
    _Float16* h_src  = (_Float16*)smem;            // 64x128
    _Float16* h_dst  = h_src + ROWS * 128;         // 64x128
    _Float16* h_edge = h_dst + ROWS * 128;         // 64x128 (activation scratch)
    _Float16* h_act  = h_edge + ROWS * 128;        // 64x256 (concat / Q|K / cto stage)
    _Float16* h_wA   = h_act + ROWS * 256;         // 128x256 weight buffer (even stages)
    _Float16* h_wB   = h_wA + 256 * 128;           // 128x128 weight buffer (odd stages)
    _Float16* h_out  = h_wB + 128 * 128;           // 64x128 (scratch / V buffer)
    float* f_m      = (float*)(h_out + ROWS * 128);
    float* f_metric = f_m      + ROWS * 16;
    float* f_sa     = f_metric + ROWS * 16;
    float* f_da     = f_sa     + ROWS * 16;
    float* f_ip     = f_da     + ROWS * 16;
    float* f_enh    = f_ip     + ROWS * 16;
    float* geo_tab  = f_enh + ROWS;                // 16x16 sign table
    float* wdg_tab  = geo_tab + 256;               // 16x16 wedge table

    const int tid   = threadIdx.x;
    const int base  = blockIdx.x * ROWS;
    const bool full = (base + ROWS) <= B;

    // ---- S0: async np weights; stage inputs; sign tables ------------------
    issue_w(ws + WS_NP, h_wA, 128 * 128);
    prefetch_g(ws, WS_TOTAL);                       // warm L2 for whole weight set
    cvt_tile(src,  h_src,  base, B);
    cvt_tile(dst,  h_dst,  base, B);
    cvt_tile(edge, h_edge, base, B);
    {
        int i = tid >> 4, j = tid & 15;             // MSB-first bit convention (N=4)
        float sign = 1.f;
#pragma unroll
        for (int k = 0; k < 4; ++k) {
            int bjk = (j >> (3 - k)) & 1;
            int pre = 0;
            for (int t = 0; t < k; ++t) pre += (i >> (3 - t)) & 1;
            if (bjk && (pre & 1)) sign = -sign;
        }
        geo_tab[tid] = sign;
        float w = 0.f;
        if ((i & j) == 0) {
            int inv = 0;
#pragma unroll
            for (int k = 0; k < 4; ++k) {
                int bik = (i >> (3 - k)) & 1;
                int suf = 0;
                for (int t = k + 1; t < 4; ++t) suf += (j >> (3 - t)) & 1;
                inv += bik * suf;
            }
            w = (inv & 1) ? -1.f : 1.f;
        }
        wdg_tab[tid] = w;
    }
    wait_w();

    // ---- S1..S5: metric-learning MLP chain, weights double-buffered -------
    issue_w(ws + WS_EP, h_wB, 128 * 128);
    gemm_ws<0, true>(h_src, 128, h_wA, 128, 128, np_b, h_act, 256, nullptr);
    wait_w();

    issue_w(ws + WS_IM1, h_wA, 256 * 128);
    gemm_ws<0, true>(h_edge, 128, h_wB, 128, 128, ep_b, h_act + 128, 256, nullptr);
    wait_w();

    issue_w(ws + WS_IM2, h_wB, 128 * 128);
    gemm_ws<1, true>(h_act, 256, h_wA, 256, 128, im1_b, h_out, 128, nullptr);
    wait_w();

    issue_w(ws + WS_MP1, h_wA, 128 * 128);
    gemm_ws<1, true>(h_out, 128, h_wB, 128, 128, im2_b, h_edge, 128, nullptr);
    wait_w();

    issue_w(ws + WS_MP2, h_wB, 16 * 128);
    gemm_ws<1, true>(h_edge, 128, h_wA, 128, 128, mp1_b, h_out, 128, nullptr);
    wait_w();

    // ---- S6: m = tanh(mp2) -------------------------------------------------
    issue_w(ws + WS_NTC, h_wA, 16 * 128);
    gemm_ws<2, false>(h_out, 128, h_wB, 128, 16, mp2_b, nullptr, 0, f_m);
    wait_w();

    // ---- S7: SPD metric (waves 0-1) + Clifford embeddings sc/dc ------------
    issue_w(ws + WS_Q, h_wB, 128 * 128);
    if (tid < ROWS) {
        const float* mm = f_m + tid * 16;
        float L[4][4];
#pragma unroll
        for (int i = 0; i < 4; ++i)
#pragma unroll
            for (int j = 0; j < 4; ++j)
                L[i][j] = (j < i) ? mm[i * 4 + j]
                        : (j == i) ? (softplus_f(mm[i * 4 + i]) + 1e-6f) : 0.f;
        float* M = f_metric + tid * 16;
#pragma unroll
        for (int i = 0; i < 4; ++i)
#pragma unroll
            for (int j = 0; j < 4; ++j) {
                float a = 0.f;
                for (int t = 0; t < 4; ++t) a += L[i][t] * L[j][t];
                M[i * 4 + j] = a;
            }
    }
    gemm_ws<0, false>(h_src, 128, h_wA, 128, 16, ntc_b, nullptr, 0, f_sa);
    gemm_ws<0, false>(h_dst, 128, h_wA, 128, 16, ntc_b, nullptr, 0, f_da);
    wait_w();

    // ---- S8: adapt + geometric/wedge/inner products (waves 0-1) + Q --------
    issue_w(ws + WS_K, h_wA, 128 * 128);
    if (tid < ROWS) {
        const int VEC[4] = {1, 2, 4, 8};
        const float* M = f_metric + tid * 16;
        float* s = f_sa + tid * 16;
        float* d = f_da + tid * 16;
        float xs[4], xd[4];
#pragma unroll
        for (int i = 0; i < 4; ++i) { xs[i] = s[VEC[i]]; xd[i] = d[VEC[i]]; }
#pragma unroll
        for (int i = 0; i < 4; ++i) {
            float as = 0.f, ad = 0.f;
            for (int j = 0; j < 4; ++j) { as += M[i * 4 + j] * xs[j]; ad += M[i * 4 + j] * xd[j]; }
            s[VEC[i]] = as; d[VEC[i]] = ad;
        }
        float gp[16], gr[16], op[16];
#pragma unroll
        for (int k = 0; k < 16; ++k) { gp[k] = 0.f; gr[k] = 0.f; op[k] = 0.f; }
        for (int i = 0; i < 16; ++i) {
            float si = s[i], di = d[i];
            for (int j = 0; j < 16; ++j) {
                int k  = i ^ j;
                float g = geo_tab[i * 16 + j];
                gp[k] += g * si * d[j];
                gr[k] += g * di * s[j];
                op[k] += wdg_tab[i * 16 + j] * si * d[j];
            }
        }
        float e = 0.f;
        float* ip = f_ip + tid * 16;
#pragma unroll
        for (int k = 0; k < 16; ++k) { ip[k] = 0.5f * (gp[k] + gr[k]); e += gp[k] + op[k]; }
        f_enh[tid] = e;
    }
    _Float16* qb = h_act;                 // 64x128
    _Float16* kb = h_act + ROWS * 128;    // 64x128
    _Float16* vb = h_out;                 // 64x128
    gemm_ws<0, true>(h_src, 128, h_wB, 128, 128, q_b, qb, 128, nullptr);
    wait_w();

    // ---- S9: K -------------------------------------------------------------
    issue_w(ws + WS_V, h_wB, 128 * 128);
    gemm_ws<0, true>(h_dst, 128, h_wA, 128, 128, k_b, kb, 128, nullptr);
    wait_w();

    // ---- S10: V, softmax weights, att = w*V --------------------------------
    issue_w(ws + WS_OUT, h_wA, 128 * 128);
    gemm_ws<0, true>(h_dst, 128, h_wB, 128, 128, v_b, vb, 128, nullptr);
    __syncthreads();                       // vb/qb/kb ready (weight copy still in flight)

    float* f_w = f_m;                      // reuse f_m as [64][8] softmax weights
    if (tid < ROWS) {
        const _Float16* q = qb + tid * 128;
        const _Float16* k = kb + tid * 128;
        float sc[8], mx = -1e30f, e = f_enh[tid];
#pragma unroll
        for (int h = 0; h < 8; ++h) {
            float a = 0.f;
            for (int dd = 0; dd < 16; ++dd) a += (float)q[h * 16 + dd] * (float)k[h * 16 + dd];
            sc[h] = a * 0.25f + e;
            mx = fmaxf(mx, sc[h]);
        }
        float sum = 0.f;
#pragma unroll
        for (int h = 0; h < 8; ++h) { sc[h] = expf(sc[h] - mx); sum += sc[h]; }
        float inv = 1.f / sum;
#pragma unroll
        for (int h = 0; h < 8; ++h) f_w[tid * 8 + h] = sc[h] * inv;
    }
    __syncthreads();

    float* f_cto = (float*)h_act;          // 16x128 f32 staged in freed Q/K space
    for (int i = tid; i < ROWS * 128; i += TPB) {
        int r = i >> 7, c = i & 127;
        h_src[i] = (_Float16)(f_w[r * 8 + (c >> 4)] * (float)vb[i]);
    }
    for (int i = tid; i < 16 * 128; i += TPB) f_cto[i] = cto_W[i];
    wait_w();                              // out_W copy + att/cto staging complete

    // ---- S11: out = att@out_W + out_b + ip@cto_W + cto_b -------------------
    gemm_final(h_src, h_wA, out_b, f_cto, cto_b, f_ip, out, base, full, B);
}

// ---------------------------------------------------------------------------
extern "C" void kernel_launch(void* const* d_in, const int* in_sizes, int n_in,
                              void* d_out, int out_size, void* d_ws, size_t ws_size,
                              hipStream_t stream) {
    (void)n_in; (void)ws_size; (void)out_size;
    const int B = in_sizes[0] / 128;
    const int grid = (B + ROWS - 1) / ROWS;
    _Float16* ws = (_Float16*)d_ws;

    // Prep: transpose+convert all WMMA weights to f16 in d_ws (L2-resident)
    auto wc = [&](const void* W, int off, int K, int N) {
        wconv_kernel<<<(K * N + TPB - 1) / TPB, TPB, 0, stream>>>(
            (const float*)W, ws + off, K, N);
    };
    wc(d_in[3],  WS_NP,  128, 128);
    wc(d_in[5],  WS_EP,  128, 128);
    wc(d_in[7],  WS_IM1, 256, 128);
    wc(d_in[9],  WS_IM2, 128, 128);
    wc(d_in[11], WS_MP1, 128, 128);
    wc(d_in[13], WS_MP2, 128, 16);
    wc(d_in[21], WS_NTC, 128, 16);
    wc(d_in[15], WS_Q,   128, 128);
    wc(d_in[17], WS_K,   128, 128);
    wc(d_in[19], WS_V,   128, 128);
    wc(d_in[25], WS_OUT, 128, 128);

    // LDS: 24576+16384+32768+16384+8192 halfs = 196608 B  + 22784 B floats = 219392 B
    const size_t smem = (size_t)(ROWS * 128 * 3 + ROWS * 256 + 256 * 128 + 128 * 128 + ROWS * 128) * 2
                      + (size_t)(ROWS * 16 * 5 + ROWS + 512) * 4;
    caga_fused_kernel<<<grid, TPB, smem, stream>>>(
        (const float*)d_in[0], (const float*)d_in[1], (const float*)d_in[2],
        ws,
        (const float*)d_in[4],  (const float*)d_in[6],  (const float*)d_in[8],
        (const float*)d_in[10], (const float*)d_in[12], (const float*)d_in[14],
        (const float*)d_in[16], (const float*)d_in[18], (const float*)d_in[20],
        (const float*)d_in[22], (const float*)d_in[23], (const float*)d_in[24],
        (const float*)d_in[26],
        (float*)d_out, B);
}
